// MyMultiHeadAttention_27771258536054
// MI455X (gfx1250) — matrix operand order repaired and checked by reading, not compiled, not run on hardware
//
#include <hip/hip_runtime.h>
#include <hip/hip_bf16.h>

// MI455X (gfx1250) fused MHA: fp16 WMMA everywhere, fp32 accumulate.
// B=4,P=4 -> BP=16 ; N=1024 ; D=512 ; H=8 ; Kd=64 ; H*Kd=512

#define BP   16
#define NN   1024
#define DD   512
#define HH   8
#define KD   64
#define HK   512

typedef __attribute__((ext_vector_type(16))) _Float16 v16h;
typedef __attribute__((ext_vector_type(8)))  _Float16 v8h;
typedef __attribute__((ext_vector_type(8)))  float    v8f;

static __device__ inline v8f wmma_f16(v16h a, v16h b, v8f c) {
    return __builtin_amdgcn_wmma_f32_16x16x32_f16(
        /*neg_a=*/false, a, /*neg_b=*/false, b,
        /*c_mod=*/(short)0, c, /*reuse_a=*/false, /*reuse_b=*/false);
}

static __device__ inline v16h make_v16(v8h lo, v8h hh) {
    v16h r;
#pragma unroll
    for (int i = 0; i < 8; ++i) { r[i] = lo[i]; r[i + 8] = hh[i]; }
    return r;
}

// A fragment (16x32 f16, MxK): lane holds row = lane&15 ; element j -> K = (j>>3)*16 + hi*8 + (j&7)
static __device__ inline v16h load_frag_a_f16(const _Float16* rowp, int kc0, int hi) {
    v8h lo = *(const v8h*)(rowp + kc0 + hi * 8);
    v8h hh = *(const v8h*)(rowp + kc0 + 16 + hi * 8);
    return make_v16(lo, hh);
}

// B fragment (32x16 f16, KxN): lane holds col = lane&15 ; element j -> K row = kc0 + (j>>3)*16 + hi*8 + (j&7)
// rowp points at the f16 row holding the K-values (same layout as A fragment).
static __device__ inline v16h load_frag_b_f16(const _Float16* rowp, int kc0, int hi) {
    const _Float16* p = rowp + kc0 + 8 * hi;
    v8h lo = *(const v8h*)(p);
    v8h hh = *(const v8h*)(p + 16);
    return make_v16(lo, hh);
}

// ---------------------------------------------------------------------------
// Prep 0: x fp32 -> f16 (16384*512 elements, 8 per thread) : 4096 blocks x 256
// ---------------------------------------------------------------------------
__global__ __launch_bounds__(256) void cvt_x_kernel(const float* __restrict__ x,
                                                    _Float16* __restrict__ xh)
{
    const size_t i = (size_t)blockIdx.x * blockDim.x + threadIdx.x;
    const float4* src = (const float4*)x;
    float4 a = src[2 * i];
    float4 b = src[2 * i + 1];
    v8h o;
    o[0] = (_Float16)a.x; o[1] = (_Float16)a.y; o[2] = (_Float16)a.z; o[3] = (_Float16)a.w;
    o[4] = (_Float16)b.x; o[5] = (_Float16)b.y; o[6] = (_Float16)b.z; o[7] = (_Float16)b.w;
    *(v8h*)(xh + 8 * i) = o;
}

// ---------------------------------------------------------------------------
// Prep 1: wq/wk/wv fp32 [512][512] -> wt f16 [w][col][512] (transposed)
// threads = 3*512*64 (8 d-values each) -> 384 blocks x 256
// ---------------------------------------------------------------------------
__global__ __launch_bounds__(256) void prep_w_kernel(
    const float* __restrict__ wq, const float* __restrict__ wk,
    const float* __restrict__ wv, _Float16* __restrict__ wt)
{
    const int t   = blockIdx.x * blockDim.x + threadIdx.x;
    const int d0  = (t & 63) * 8;
    const int col = (t >> 6) & 511;
    const int w   = t >> 15;
    const float* W = (w == 0) ? wq : (w == 1) ? wk : wv;
    v8h o;
#pragma unroll
    for (int j = 0; j < 8; ++j) o[j] = (_Float16)W[(size_t)(d0 + j) * HK + col];
    *(v8h*)(wt + ((size_t)w * HK + col) * DD + d0) = o;
}

// ---------------------------------------------------------------------------
// Prep 2: wo fp32 [512][64] -> wot f16 [64][512] (transposed)
// threads = 64*64 -> 16 blocks x 256
// ---------------------------------------------------------------------------
__global__ __launch_bounds__(256) void prep_wo_kernel(const float* __restrict__ wo,
                                                      _Float16* __restrict__ wot)
{
    const int t   = blockIdx.x * blockDim.x + threadIdx.x;
    const int d0  = (t & 63) * 8;
    const int col = t >> 6;   // 0..63
    v8h o;
#pragma unroll
    for (int j = 0; j < 8; ++j) o[j] = (_Float16)wo[(size_t)(d0 + j) * KD + col];
    *(v8h*)(wot + (size_t)col * HK + d0) = o;
}

// ---------------------------------------------------------------------------
// Kernel 1: QKV projection. Wave computes 64 rows x 64 cols (one head) of one
// of {Q,K,V}. Q gets 1/sqrt(Kd) folded in; V is stored transposed [h][64][N].
// All selector indices (w, head, bp) are block-uniform -> scalar branches.
// grid = 768 blocks x 256 threads :
//   rb = (blockIdx%2)*8 + wave_in_block ; cb = (blockIdx/2)%8 ;
//   w  = (blockIdx/16)%3 ; bp = blockIdx/48
// ---------------------------------------------------------------------------
__global__ __launch_bounds__(256) void qkv_kernel(
    const _Float16* __restrict__ xh, const _Float16* __restrict__ wt,
    const float* __restrict__ bq, const float* __restrict__ bk,
    const float* __restrict__ bv,
    _Float16* __restrict__ qws, _Float16* __restrict__ kws,
    _Float16* __restrict__ vtws)
{
    const int wib  = threadIdx.x >> 5;     // wave in block (0..7)
    const int lane = threadIdx.x & 31;
    const int hi   = lane >> 4;
    const int l16  = lane & 15;

    const int blk = blockIdx.x;
    const int rb  = (blk & 1) * 8 + wib;   // row block (64 rows)
    const int cb  = (blk >> 1) & 7;        // col block == head (scalar)
    const int w   = (blk >> 4) % 3;        // 0=q 1=k 2=v        (scalar)
    const int bp  = blk / 48;              //                    (scalar)

    const _Float16* Wt   = wt + (size_t)w * HK * DD;
    const float*    bias = (w == 0) ? bq : (w == 1) ? bk : bv;
    const _Float16* xb   = xh + (size_t)bp * NN * DD;

    const int n0 = rb * 64;
    const int c0 = cb * 64;

    v8f acc[4][4];
#pragma unroll
    for (int rt = 0; rt < 4; ++rt)
#pragma unroll
        for (int ct = 0; ct < 4; ++ct) acc[rt][ct] = (v8f)0.0f;

    for (int kc0 = 0; kc0 < DD; kc0 += 32) {
        v16h A[4];
#pragma unroll
        for (int rt = 0; rt < 4; ++rt)
            A[rt] = load_frag_a_f16(xb + (size_t)(n0 + rt * 16 + l16) * DD, kc0, hi);
        v16h Bm[4];
#pragma unroll
        for (int ct = 0; ct < 4; ++ct)
            Bm[ct] = load_frag_b_f16(Wt + (size_t)(c0 + ct * 16 + l16) * DD, kc0, hi);
#pragma unroll
        for (int rt = 0; rt < 4; ++rt)
#pragma unroll
            for (int ct = 0; ct < 4; ++ct)
                acc[rt][ct] = wmma_f16(A[rt], Bm[ct], acc[rt][ct]);
    }

    const int h = cb;
    if (w == 2) {
        // V: store transposed [bp][h][k][n]
        _Float16* vbase = vtws + ((size_t)bp * HH + h) * KD * NN;
#pragma unroll
        for (int rt = 0; rt < 4; ++rt)
#pragma unroll
            for (int ct = 0; ct < 4; ++ct) {
                const int   kk   = ct * 16 + l16;
                const float bval = bias[c0 + kk];
#pragma unroll
                for (int e = 0; e < 8; ++e) {
                    const int row = n0 + rt * 16 + e + 8 * hi;
                    vbase[(size_t)kk * NN + row] = (_Float16)(acc[rt][ct][e] + bval);
                }
            }
    } else {
        // Q or K: store [bp][h][n][k] ; Q gets 1/sqrt(64) folded in
        _Float16* qkbase = ((w == 0) ? qws : kws) + ((size_t)bp * HH + h) * NN * KD;
        const float scale = (w == 0) ? 0.125f : 1.0f;
#pragma unroll
        for (int rt = 0; rt < 4; ++rt)
#pragma unroll
            for (int ct = 0; ct < 4; ++ct) {
                const int   kk   = ct * 16 + l16;
                const float bval = bias[c0 + kk];
#pragma unroll
                for (int e = 0; e < 8; ++e) {
                    const int row = n0 + rt * 16 + e + 8 * hi;
                    qkbase[(size_t)row * KD + kk] = (_Float16)((acc[rt][ct][e] + bval) * scale);
                }
            }
    }
}

// ---------------------------------------------------------------------------
// Kernel 2: flash attention per (bp, head, 64-row block). 4 waves x 16 rows.
// grid = BP * H * (N/64) = 2048 blocks x 128 threads
// ---------------------------------------------------------------------------
__global__ __launch_bounds__(128) void attn_kernel(
    const _Float16* __restrict__ qws, const _Float16* __restrict__ kws,
    const _Float16* __restrict__ vtws, _Float16* __restrict__ ctx)
{
    __shared__ _Float16 plds[4 * 16 * 72];   // per-wave 16x64 P tile, padded rows (72 halves)

    const int tid  = threadIdx.x;
    const int wave = tid >> 5;
    const int lane = tid & 31;
    const int hi   = lane >> 4;
    const int l16  = lane & 15;

    const int rb = blockIdx.x & 15;
    const int h  = (blockIdx.x >> 4) & 7;
    const int bp = blockIdx.x >> 7;

    const size_t head = (size_t)bp * HH + h;
    const _Float16* Qb  = qws  + head * NN * KD;
    const _Float16* Kb  = kws  + head * NN * KD;
    const _Float16* Vtb = vtws + head * KD * NN;

    const int n0 = rb * 64 + wave * 16;   // this wave's 16 query rows

    v16h Qf[2];
#pragma unroll
    for (int kc = 0; kc < 2; ++kc)
        Qf[kc] = load_frag_a_f16(Qb + (size_t)(n0 + l16) * KD, kc * 32, hi);

    float m_run[8], s_run[8];
    v8f   O[4];
#pragma unroll
    for (int e = 0; e < 8; ++e) { m_run[e] = -3.0e38f; s_run[e] = 0.0f; }
#pragma unroll
    for (int ct = 0; ct < 4; ++ct) O[ct] = (v8f)0.0f;

    _Float16* myrow = plds + (wave * 16 + l16) * 72;  // A-frag read base (row = l16)

    for (int kb = 0; kb < NN / 64; ++kb) {
        const int key0 = kb * 64;

        // S = (Q*scale) . K^T  :  4 col tiles x 2 k-steps
        v8f S[4];
#pragma unroll
        for (int ct = 0; ct < 4; ++ct) S[ct] = (v8f)0.0f;
#pragma unroll
        for (int ct = 0; ct < 4; ++ct) {
            const _Float16* krow = Kb + (size_t)(key0 + ct * 16 + l16) * KD;
#pragma unroll
            for (int kc = 0; kc < 2; ++kc) {
                v16h Bf = load_frag_b_f16(krow, kc * 32, hi);
                S[ct] = wmma_f16(Qf[kc], Bf, S[ct]);
            }
        }

        // online softmax per row (row = e + 8*hi, cols striped over 16 lanes x 4 tiles)
#pragma unroll
        for (int e = 0; e < 8; ++e) {
            float v = fmaxf(fmaxf(S[0][e], S[1][e]), fmaxf(S[2][e], S[3][e]));
            v = fmaxf(v, __shfl_xor(v, 1, 32));
            v = fmaxf(v, __shfl_xor(v, 2, 32));
            v = fmaxf(v, __shfl_xor(v, 4, 32));
            v = fmaxf(v, __shfl_xor(v, 8, 32));
            const float mn    = fmaxf(m_run[e], v);
            const float alpha = __expf(m_run[e] - mn);
            m_run[e] = mn;
#pragma unroll
            for (int ct = 0; ct < 4; ++ct) O[ct][e] *= alpha;
            float ps = 0.0f;
#pragma unroll
            for (int ct = 0; ct < 4; ++ct) {
                const float p = __expf(S[ct][e] - mn);
                S[ct][e] = p;
                ps += p;
            }
            ps += __shfl_xor(ps, 1, 32);
            ps += __shfl_xor(ps, 2, 32);
            ps += __shfl_xor(ps, 4, 32);
            ps += __shfl_xor(ps, 8, 32);
            s_run[e] = s_run[e] * alpha + ps;
        }

        // P (C-layout) -> LDS -> A-layout fragments
#pragma unroll
        for (int ct = 0; ct < 4; ++ct)
#pragma unroll
            for (int e = 0; e < 8; ++e)
                plds[(wave * 16 + e + 8 * hi) * 72 + ct * 16 + l16] = (_Float16)S[ct][e];
        __syncthreads();
        v16h Pf[2];
#pragma unroll
        for (int kc = 0; kc < 2; ++kc)
            Pf[kc] = load_frag_a_f16(myrow, kc * 32, hi);
        __syncthreads();

        // O += P . V   (V stored transposed -> contiguous B-frag loads)
#pragma unroll
        for (int ct = 0; ct < 4; ++ct) {
            const _Float16* vrow = Vtb + (size_t)(ct * 16 + l16) * NN + key0;
#pragma unroll
            for (int kc = 0; kc < 2; ++kc) {
                v16h Bf = load_frag_b_f16(vrow, kc * 32, hi);
                O[ct] = wmma_f16(Pf[kc], Bf, O[ct]);
            }
        }
    }

    // normalize and write ctx [bp][n][h*64 + k] in f16
#pragma unroll
    for (int ct = 0; ct < 4; ++ct)
#pragma unroll
        for (int e = 0; e < 8; ++e) {
            const int row = n0 + e + 8 * hi;
            const float v = O[ct][e] / s_run[e];
            ctx[((size_t)bp * NN + row) * HK + h * KD + ct * 16 + l16] = (_Float16)v;
        }
}

// ---------------------------------------------------------------------------
// Kernel 3: output projection  ctx[16384x512] f16  x  wot[64][512] f16 -> out f32
// wave computes 64 rows x 64 cols ; waves = 16384/64 = 256 -> 32 blocks x 256
// ---------------------------------------------------------------------------
__global__ __launch_bounds__(256) void proj_kernel(
    const _Float16* __restrict__ ctx, const _Float16* __restrict__ wot,
    const float* __restrict__ bo, float* __restrict__ out)
{
    const int gid  = blockIdx.x * blockDim.x + threadIdx.x;
    const int wave = gid >> 5;
    const int lane = threadIdx.x & 31;
    const int hi   = lane >> 4;
    const int l16  = lane & 15;
    const int n0   = wave * 64;

    v8f acc[4][4];
#pragma unroll
    for (int rt = 0; rt < 4; ++rt)
#pragma unroll
        for (int ct = 0; ct < 4; ++ct) acc[rt][ct] = (v8f)0.0f;

    for (int kc0 = 0; kc0 < HK; kc0 += 32) {
        v16h A[4];
#pragma unroll
        for (int rt = 0; rt < 4; ++rt)
            A[rt] = load_frag_a_f16(ctx + (size_t)(n0 + rt * 16 + l16) * HK, kc0, hi);
        v16h Bm[4];
#pragma unroll
        for (int ct = 0; ct < 4; ++ct)
            Bm[ct] = load_frag_b_f16(wot + (size_t)(ct * 16 + l16) * HK, kc0, hi);
#pragma unroll
        for (int rt = 0; rt < 4; ++rt)
#pragma unroll
            for (int ct = 0; ct < 4; ++ct)
                acc[rt][ct] = wmma_f16(A[rt], Bm[ct], acc[rt][ct]);
    }

#pragma unroll
    for (int rt = 0; rt < 4; ++rt)
#pragma unroll
        for (int ct = 0; ct < 4; ++ct) {
            const float bval = bo[ct * 16 + l16];
#pragma unroll
            for (int e = 0; e < 8; ++e) {
                const int row = n0 + rt * 16 + e + 8 * hi;
                out[(size_t)row * KD + ct * 16 + l16] = acc[rt][ct][e] + bval;
            }
        }
}

// ---------------------------------------------------------------------------
extern "C" void kernel_launch(void* const* d_in, const int* in_sizes, int n_in,
                              void* d_out, int out_size, void* d_ws, size_t ws_size,
                              hipStream_t stream) {
    const float* x  = (const float*)d_in[0];
    const float* wq = (const float*)d_in[1];
    const float* bq = (const float*)d_in[2];
    const float* wk = (const float*)d_in[3];
    const float* bk = (const float*)d_in[4];
    const float* wv = (const float*)d_in[5];
    const float* bv = (const float*)d_in[6];
    const float* wo = (const float*)d_in[7];
    const float* bo = (const float*)d_in[8];
    float* out = (float*)d_out;

    // workspace (halves): Q | K | V^T | ctx | xh | wt | wot
    _Float16* ws   = (_Float16*)d_ws;
    const size_t seg = (size_t)BP * HH * NN * KD;   // 8,388,608 halves
    _Float16* qws  = ws;
    _Float16* kws  = ws + seg;
    _Float16* vtws = ws + 2 * seg;
    _Float16* ctxw = ws + 3 * seg;
    _Float16* xh   = ws + 4 * seg;                  // BP*NN*DD = 8,388,608 halves
    _Float16* wt   = ws + 5 * seg;                  // 3*HK*DD  =   786,432 halves
    _Float16* wot  = wt + (size_t)3 * HK * DD;      // KD*HK    =    32,768 halves

    // prep (independent of each other)
    cvt_x_kernel<<<4096, 256, 0, stream>>>(x, xh);
    prep_w_kernel<<<384, 256, 0, stream>>>(wq, wk, wv, wt);
    prep_wo_kernel<<<16, 256, 0, stream>>>(wo, wot);

    // 768 blocks of 256 (8 waves each)
    qkv_kernel<<<768, 256, 0, stream>>>(xh, wt, bq, bk, bv, qws, kws, vtws);
    // 16 bp * 8 heads * 16 row-blocks = 2048 blocks of 128
    attn_kernel<<<2048, 128, 0, stream>>>(qws, kws, vtws, ctxw);
    // 256 waves -> 32 blocks of 256
    proj_kernel<<<32, 256, 0, stream>>>(ctxw, wot, bo, out);
}
